// CrossAttentionFusion_41652592836886
// MI455X (gfx1250) — 16-bit and 8-bit matrix operand order repaired and checked by reading, compiled, not run on hardware
//
#include <hip/hip_runtime.h>

typedef __attribute__((ext_vector_type(16))) _Float16 v16h;
typedef __attribute__((ext_vector_type(8)))  _Float16 v8h;
typedef __attribute__((ext_vector_type(8)))  float    v8f;
typedef __attribute__((ext_vector_type(8)))  int      v8i;
typedef __attribute__((ext_vector_type(4)))  int      v4i;
typedef __attribute__((ext_vector_type(2)))  int      v2i;

constexpr int Bb   = 4;
constexpr int Ss   = 2048;
constexpr int Dsem = 1024;
constexpr int Dpro = 512;
constexpr int Hh   = 8;
constexpr int Hd   = 128;
constexpr int Dff  = 4 * Dsem;
constexpr int Mtok = Bb * Ss;
constexpr float kQKScale = 0.08838834764831845f;


__device__ __forceinline__ v16h frag_a_f16(const _Float16* base, int ld) {
  int lane = threadIdx.x & 31;
  const _Float16* p = base + (size_t)(lane & 15) * ld + ((lane >> 4) << 3);
  v8h lo = *(const v8h*)(p);
  v8h hi = *(const v8h*)(p + 16);
  v16h f;
#pragma unroll
  for (int i = 0; i < 8; ++i) { f[i] = lo[i]; f[i + 8] = hi[i]; }
  return f;
}

__device__ __forceinline__ v16h frag_b_f16(const _Float16* base, int ld) {
  int lane = threadIdx.x & 31;
  const _Float16* p = base + (size_t)(lane & 15) * ld + ((lane >> 4) << 3);
  v8h lo = *(const v8h*)(p);
  v8h hi = *(const v8h*)(p + 16);
  v16h f;
#pragma unroll
  for (int i = 0; i < 8; ++i) { f[i] = lo[i]; f[i + 8] = hi[i]; }
  return f;
}

__device__ __forceinline__ v8i frag_a_i8(const signed char* base, int ld) {
  int lane = threadIdx.x & 31;
  const signed char* p = base + (size_t)(lane & 15) * ld + ((lane >> 4) << 4);
  v4i c0 = *(const v4i*)(p);
  v4i c1 = *(const v4i*)(p + 32);
  v8i f;
  f[0] = c0[0]; f[1] = c0[1]; f[2] = c0[2]; f[3] = c0[3];
  f[4] = c1[0]; f[5] = c1[1]; f[6] = c1[2]; f[7] = c1[3];
  return f;
}

__device__ __forceinline__ v8i frag_b_i8(const signed char* base, int ld) {
  int lane = threadIdx.x & 31;
  const signed char* p = base + (size_t)(lane & 15) * ld + ((lane >> 4) << 4);
  v4i c0 = *(const v4i*)(p);
  v4i c1 = *(const v4i*)(p + 32);
  v8i f;
  f[0] = c0[0]; f[1] = c0[1]; f[2] = c0[2]; f[3] = c0[3];
  f[4] = c1[0]; f[5] = c1[1]; f[6] = c1[2]; f[7] = c1[3];
  return f;
}

__device__ __forceinline__ v8f zero8f() {
  v8f z;
#pragma unroll
  for (int i = 0; i < 8; ++i) z[i] = 0.0f;
  return z;
}
__device__ __forceinline__ v8i zero8i() {
  v8i z;
#pragma unroll
  for (int i = 0; i < 8; ++i) z[i] = 0;
  return z;
}

__device__ __forceinline__ float block_sum(float v, float* sred) {
  int tid = threadIdx.x;
#pragma unroll
  for (int m = 16; m; m >>= 1) v += __shfl_xor(v, m);
  __syncthreads();
  if ((tid & 31) == 0) sred[tid >> 5] = v;
  __syncthreads();
  if (tid == 0) {
    int nw = blockDim.x >> 5;
    float s = sred[0];
    for (int i = 1; i < nw; ++i) s += sred[i];
    sred[0] = s;
  }
  __syncthreads();
  return sred[0];
}
__device__ __forceinline__ float block_max(float v, float* sred) {
  int tid = threadIdx.x;
#pragma unroll
  for (int m = 16; m; m >>= 1) v = fmaxf(v, __shfl_xor(v, m));
  __syncthreads();
  if ((tid & 31) == 0) sred[tid >> 5] = v;
  __syncthreads();
  if (tid == 0) {
    int nw = blockDim.x >> 5;
    float s = sred[0];
    for (int i = 1; i < nw; ++i) s = fmaxf(s, sred[i]);
    sred[0] = s;
  }
  __syncthreads();
  return sred[0];
}

__global__ void rmsnorm_f16_kernel(const float* __restrict__ src, const float* __restrict__ g,
                                   _Float16* __restrict__ dst, int D) {
  __shared__ float sred[32];
  int row = blockIdx.x, tid = threadIdx.x;
  const float* x = src + (size_t)row * D;
  float ss = 0.0f;
  for (int i = tid; i < D; i += blockDim.x) { float v = x[i]; ss += v * v; }
  ss = block_sum(ss, sred);
  float rstd = rsqrtf(ss / (float)D + 1e-6f);
  _Float16* y = dst + (size_t)row * D;
  for (int i = tid; i < D; i += blockDim.x) y[i] = (_Float16)(x[i] * rstd * g[i]);
}

__global__ void f32_to_f16_kernel(const float* __restrict__ src, _Float16* __restrict__ dst, int n) {
  int i = blockIdx.x * blockDim.x + threadIdx.x;
  if (i < n) dst[i] = (_Float16)src[i];
}

__global__ void xquant_kernel(const float* __restrict__ sem2, const float* __restrict__ g,
                              signed char* __restrict__ qx, float* __restrict__ xscale) {
  __shared__ float xs[Dsem];
  __shared__ float sred[32];
  int row = blockIdx.x, tid = threadIdx.x;
  const float* x = sem2 + (size_t)row * Dsem;
  float ss = 0.0f;
  for (int i = tid; i < Dsem; i += blockDim.x) { float v = x[i]; ss += v * v; }
  ss = block_sum(ss, sred);
  float rstd = rsqrtf(ss / (float)Dsem + 1e-6f);
  float amax = 0.0f;
  for (int i = tid; i < Dsem; i += blockDim.x) {
    float v = x[i] * rstd * g[i];
    xs[i] = v;
    amax = fmaxf(amax, fabsf(v));
  }
  amax = fmaxf(block_max(amax, sred), 1e-5f);
  float s = 127.0f / amax;
  for (int i = tid; i < Dsem; i += blockDim.x) {
    float q = rintf(xs[i] * s);
    q = fminf(fmaxf(q, -128.0f), 127.0f);
    qx[(size_t)row * Dsem + i] = (signed char)q;
  }
  if (tid == 0) xscale[row] = amax / 127.0f;
}

__global__ void hquant_kernel(const float* __restrict__ hbuf, signed char* __restrict__ qh,
                              float* __restrict__ hscale) {
  __shared__ float sred[32];
  int row = blockIdx.x, tid = threadIdx.x;
  const float* x = hbuf + (size_t)row * Dff;
  float amax = 0.0f;
  for (int i = tid; i < Dff; i += blockDim.x) amax = fmaxf(amax, fabsf(x[i]));
  amax = fmaxf(block_max(amax, sred), 1e-5f);
  float s = 127.0f / amax;
  for (int i = tid; i < Dff; i += blockDim.x) {
    float q = rintf(x[i] * s);
    q = fminf(fmaxf(q, -128.0f), 127.0f);
    qh[(size_t)row * Dff + i] = (signed char)q;
  }
  if (tid == 0) hscale[row] = amax / 127.0f;
}

__global__ void absmean_partial_kernel(const float* __restrict__ w, int n, float* __restrict__ partial) {
  __shared__ float sred[32];
  float s = 0.0f;
  for (int i = blockIdx.x * blockDim.x + threadIdx.x; i < n; i += gridDim.x * blockDim.x)
    s += fabsf(w[i]);
  s = block_sum(s, sred);
  if (threadIdx.x == 0) partial[blockIdx.x] = s;
}
__global__ void absmean_final_kernel(const float* __restrict__ partial, int nb, int n,
                                     float* __restrict__ out) {
  __shared__ float sred[32];
  float s = 0.0f;
  for (int i = threadIdx.x; i < nb; i += blockDim.x) s += partial[i];
  s = block_sum(s, sred);
  if (threadIdx.x == 0) out[0] = fmaxf(s / (float)n, 1e-5f);
}

__global__ void wquant_kernel(const float* __restrict__ w, const float* __restrict__ wscale,
                              signed char* __restrict__ qw, int n) {
  int i = blockIdx.x * blockDim.x + threadIdx.x;
  if (i < n) {
    float s = 1.0f / wscale[0];
    float q = rintf(w[i] * s);
    q = fminf(fmaxf(q, -1.0f), 1.0f);
    qw[i] = (signed char)q;
  }
}

__global__ __launch_bounds__(256) void gemm_f16_kernel(
    const _Float16* __restrict__ A, const _Float16* __restrict__ W,
    const float* __restrict__ bias, _Float16* __restrict__ out,
    int N, int K, int vmode) {
  int wid  = threadIdx.x >> 5;
  int gw   = blockIdx.x * (blockDim.x >> 5) + wid;
  int NT   = N >> 6;
  int mt   = gw / NT, nt = gw % NT;
  int lane = threadIdx.x & 31;

  v8f acc[4];
#pragma unroll
  for (int j = 0; j < 4; ++j) acc[j] = zero8f();

  for (int k0 = 0; k0 < K; k0 += 32) {
    v16h a = frag_a_f16(A + (size_t)(mt * 16) * K + k0, K);
#pragma unroll
    for (int j = 0; j < 4; ++j) {
      v16h b = frag_b_f16(W + (size_t)(nt * 64 + j * 16) * K + k0, K);
      acc[j] = __builtin_amdgcn_wmma_f32_16x16x32_f16(false, a, false, b, (short)0, acc[j],
                                                      false, false);
    }
  }

  int colr = lane & 15, rowoff = (lane >> 4) << 3;
  int mbase = mt * 16 + rowoff;
  if (!vmode) {
#pragma unroll
    for (int j = 0; j < 4; ++j) {
      int n = nt * 64 + j * 16 + colr;
      float bb = bias[n];
#pragma unroll
      for (int r = 0; r < 8; ++r)
        out[(size_t)(mbase + r) * N + n] = (_Float16)(acc[j][r] + bb);
    }
  } else {
    int b = mbase >> 11;
    int sbase = mbase & (Ss - 1);
#pragma unroll
    for (int j = 0; j < 4; ++j) {
      int n = nt * 64 + j * 16 + colr;
      int h = n >> 7, d = n & (Hd - 1);
      float bb = bias[n];
      v8h pk;
#pragma unroll
      for (int r = 0; r < 8; ++r) pk[r] = (_Float16)(acc[j][r] + bb);
      *(v8h*)(out + ((size_t)(b * Hh + h) * Hd + d) * Ss + sbase) = pk;
    }
  }
}

__global__ __launch_bounds__(256) void gemm_o_kernel(
    const _Float16* __restrict__ A, const _Float16* __restrict__ W,
    const float* __restrict__ bias, const float* __restrict__ res,
    float* __restrict__ out) {
  const int N = Dsem, K = Dsem;
  int wid  = threadIdx.x >> 5;
  int gw   = blockIdx.x * (blockDim.x >> 5) + wid;
  int NT   = N >> 6;
  int mt   = gw / NT, nt = gw % NT;
  int lane = threadIdx.x & 31;

  v8f acc[4];
#pragma unroll
  for (int j = 0; j < 4; ++j) acc[j] = zero8f();

  for (int k0 = 0; k0 < K; k0 += 32) {
    v16h a = frag_a_f16(A + (size_t)(mt * 16) * K + k0, K);
#pragma unroll
    for (int j = 0; j < 4; ++j) {
      v16h b = frag_b_f16(W + (size_t)(nt * 64 + j * 16) * K + k0, K);
      acc[j] = __builtin_amdgcn_wmma_f32_16x16x32_f16(false, a, false, b, (short)0, acc[j],
                                                      false, false);
    }
  }
  int colr = lane & 15, rowoff = (lane >> 4) << 3;
  int mbase = mt * 16 + rowoff;
#pragma unroll
  for (int j = 0; j < 4; ++j) {
    int n = nt * 64 + j * 16 + colr;
    float bb = bias[n];
#pragma unroll
    for (int r = 0; r < 8; ++r) {
      size_t idx = (size_t)(mbase + r) * N + n;
      out[idx] = res[idx] + acc[j][r] + bb;
    }
  }
}

__global__ __launch_bounds__(256) void attn_kernel(
    const _Float16* __restrict__ Q, const _Float16* __restrict__ K,
    const _Float16* __restrict__ Vt, _Float16* __restrict__ ctxOut) {
  __shared__ float sm[8][16][34];
  int wid  = threadIdx.x >> 5;
  int lane = threadIdx.x & 31;
  int gw   = blockIdx.x * (blockDim.x >> 5) + wid;
  int qt = gw & ((Ss / 16) - 1);
  int bh = gw >> 7;
  int h  = bh & (Hh - 1);
  int b  = bh >> 3;

  const _Float16* Qbase = Q  + (size_t)(b * Ss + qt * 16) * Dsem + h * Hd;
  const _Float16* Kbase = K  + (size_t)(b * Ss) * Dsem + h * Hd;
  const _Float16* Vbase = Vt + (size_t)(b * Hh + h) * Hd * Ss;

  v16h qf[4];
#pragma unroll
  for (int df = 0; df < 4; ++df) {
    qf[df] = frag_a_f16(Qbase + df * 32, Dsem);
#pragma unroll
    for (int i = 0; i < 16; ++i) qf[df][i] = (_Float16)((float)qf[df][i] * kQKScale);
  }

  v8f ctx[8];
#pragma unroll
  for (int dt = 0; dt < 8; ++dt) ctx[dt] = zero8f();
  float mrow[8], lrow[8];
#pragma unroll
  for (int r = 0; r < 8; ++r) { mrow[r] = -3.0e38f; lrow[r] = 0.0f; }

  int colr = lane & 15, rowoff = (lane >> 4) << 3;

  for (int kb = 0; kb < Ss; kb += 32) {
    v8f s0 = zero8f(), s1 = zero8f();
#pragma unroll
    for (int df = 0; df < 4; ++df) {
      v16h k0 = frag_b_f16(Kbase + (size_t)kb * Dsem + df * 32, Dsem);
      v16h k1 = frag_b_f16(Kbase + (size_t)(kb + 16) * Dsem + df * 32, Dsem);
      s0 = __builtin_amdgcn_wmma_f32_16x16x32_f16(false, qf[df], false, k0, (short)0, s0,
                                                  false, false);
      s1 = __builtin_amdgcn_wmma_f32_16x16x32_f16(false, qf[df], false, k1, (short)0, s1,
                                                  false, false);
    }
    float p0[8], p1[8];
#pragma unroll
    for (int r = 0; r < 8; ++r) {
      float mx = fmaxf(s0[r], s1[r]);
      mx = fmaxf(mx, __shfl_xor(mx, 1));
      mx = fmaxf(mx, __shfl_xor(mx, 2));
      mx = fmaxf(mx, __shfl_xor(mx, 4));
      mx = fmaxf(mx, __shfl_xor(mx, 8));
      float mnew = fmaxf(mrow[r], mx);
      float sc = __expf(mrow[r] - mnew);
      mrow[r] = mnew;
      p0[r] = __expf(s0[r] - mnew);
      p1[r] = __expf(s1[r] - mnew);
      float la = p0[r] + p1[r];
      la += __shfl_xor(la, 1);
      la += __shfl_xor(la, 2);
      la += __shfl_xor(la, 4);
      la += __shfl_xor(la, 8);
      lrow[r] = lrow[r] * sc + la;
#pragma unroll
      for (int dt = 0; dt < 8; ++dt) ctx[dt][r] *= sc;
    }
#pragma unroll
    for (int r = 0; r < 8; ++r) {
      sm[wid][r + rowoff][colr]      = p0[r];
      sm[wid][r + rowoff][colr + 16] = p1[r];
    }
    asm volatile("s_wait_dscnt 0x0" ::: "memory");
    v16h pf;
    {
      int pr = lane & 15, pkb = (lane >> 4) << 3;
#pragma unroll
      for (int i = 0; i < 8; ++i) {
        pf[i]     = (_Float16)sm[wid][pr][pkb + i];
        pf[i + 8] = (_Float16)sm[wid][pr][pkb + 16 + i];
      }
    }
    asm volatile("s_wait_dscnt 0x0" ::: "memory");
#pragma unroll
    for (int dt = 0; dt < 8; ++dt) {
      v16h vf = frag_b_f16(Vbase + (size_t)(dt * 16) * Ss + kb, Ss);
      ctx[dt] = __builtin_amdgcn_wmma_f32_16x16x32_f16(false, pf, false, vf, (short)0, ctx[dt],
                                                       false, false);
    }
  }

  _Float16* ob = ctxOut + (size_t)(b * Ss + qt * 16) * Dsem + h * Hd;
#pragma unroll
  for (int dt = 0; dt < 8; ++dt) {
#pragma unroll
    for (int r = 0; r < 8; ++r)
      ob[(size_t)(r + rowoff) * Dsem + dt * 16 + colr] = (_Float16)(ctx[dt][r] / lrow[r]);
  }
}

__global__ __launch_bounds__(256) void ffn1_kernel(
    const signed char* __restrict__ qx, const signed char* __restrict__ qw1,
    const float* __restrict__ xscale, const float* __restrict__ w1scale,
    const float* __restrict__ alpha, const float* __restrict__ beta,
    float* __restrict__ hbuf) {
  int wid  = threadIdx.x >> 5;
  int gw   = blockIdx.x * (blockDim.x >> 5) + wid;
  int NT   = Dff >> 6;
  int mt   = gw / NT, nt = gw % NT;
  int lane = threadIdx.x & 31;

  v8i acc[4];
#pragma unroll
  for (int j = 0; j < 4; ++j) acc[j] = zero8i();

  for (int k0 = 0; k0 < Dsem; k0 += 64) {
    v8i a = frag_a_i8(qx + (size_t)(mt * 16) * Dsem + k0, Dsem);
#pragma unroll
    for (int j = 0; j < 4; ++j) {
      v8i bw = frag_b_i8(qw1 + (size_t)(nt * 64 + j * 16) * Dsem + k0, Dsem);
      acc[j] = __builtin_amdgcn_wmma_i32_16x16x64_iu8(true, a, true, bw, acc[j], false, false);
    }
  }
  float ws = w1scale[0];
  int colr = lane & 15, rowoff = (lane >> 4) << 3;
  int mbase = mt * 16 + rowoff;
#pragma unroll
  for (int j = 0; j < 4; ++j) {
    int n = nt * 64 + j * 16 + colr;
    float al = alpha[n];
    float ib = 1.0f / (beta[n] + 1e-9f);
#pragma unroll
    for (int r = 0; r < 8; ++r) {
      int m = mbase + r;
      float hv = (float)acc[j][r] * xscale[m] * ws;
      float sn = __sinf(al * hv);
      hbuf[(size_t)m * Dff + n] = hv + ib * sn * sn;
    }
  }
}

__global__ __launch_bounds__(256) void ffn2_kernel(
    const signed char* __restrict__ qh, const signed char* __restrict__ qw2,
    const float* __restrict__ hscale, const float* __restrict__ w2scale,
    const float* __restrict__ sem2, float* __restrict__ outp) {
  int wid  = threadIdx.x >> 5;
  int gw   = blockIdx.x * (blockDim.x >> 5) + wid;
  int NT   = Dsem >> 6;
  int mt   = gw / NT, nt = gw % NT;
  int lane = threadIdx.x & 31;

  v8i acc[4];
#pragma unroll
  for (int j = 0; j < 4; ++j) acc[j] = zero8i();

  for (int k0 = 0; k0 < Dff; k0 += 64) {
    v8i a = frag_a_i8(qh + (size_t)(mt * 16) * Dff + k0, Dff);
#pragma unroll
    for (int j = 0; j < 4; ++j) {
      v8i bw = frag_b_i8(qw2 + (size_t)(nt * 64 + j * 16) * Dff + k0, Dff);
      acc[j] = __builtin_amdgcn_wmma_i32_16x16x64_iu8(true, a, true, bw, acc[j], false, false);
    }
  }
  float ws = w2scale[0];
  int colr = lane & 15, rowoff = (lane >> 4) << 3;
  int mbase = mt * 16 + rowoff;
#pragma unroll
  for (int j = 0; j < 4; ++j) {
    int n = nt * 64 + j * 16 + colr;
#pragma unroll
    for (int r = 0; r < 8; ++r) {
      int m = mbase + r;
      size_t idx = (size_t)m * Dsem + n;
      outp[idx] = sem2[idx] + (float)acc[j][r] * hscale[m] * ws;
    }
  }
}

extern "C" void kernel_launch(void* const* d_in, const int* in_sizes, int n_in,
                              void* d_out, int out_size, void* d_ws, size_t ws_size,
                              hipStream_t stream) {
  (void)in_sizes; (void)n_in; (void)out_size; (void)ws_size;
  const float* sem   = (const float*)d_in[0];
  const float* pro   = (const float*)d_in[1];
  const float* g_sem = (const float*)d_in[2];
  const float* g_pro = (const float*)d_in[3];
  const float* g_ff  = (const float*)d_in[4];
  const float* Wq = (const float*)d_in[5];  const float* bq = (const float*)d_in[6];
  const float* Wk = (const float*)d_in[7];  const float* bk = (const float*)d_in[8];
  const float* Wv = (const float*)d_in[9];  const float* bv = (const float*)d_in[10];
  const float* Wo = (const float*)d_in[11]; const float* bo = (const float*)d_in[12];
  const float* W1 = (const float*)d_in[13];
  const float* W2 = (const float*)d_in[14];
  const float* alpha = (const float*)d_in[15];
  const float* beta  = (const float*)d_in[16];
  float* outp = (float*)d_out;

  char* wsb = (char*)d_ws;
  size_t cur = 0;
  auto alloc = [&](size_t bytes) -> char* {
    char* p = wsb + cur;
    cur += (bytes + 255) & ~(size_t)255;
    return p;
  };
  float*       sem2   = (float*)alloc((size_t)Mtok * Dsem * 4);
  signed char* qx     = (signed char*)alloc((size_t)Mtok * Dsem);
  float*       xscale = (float*)alloc((size_t)Mtok * 4);
  signed char* qw1    = (signed char*)alloc((size_t)Dff * Dsem);
  signed char* qw2    = (signed char*)alloc((size_t)Dsem * Dff);
  float*       w1s    = (float*)alloc(256);
  float*       w2s    = (float*)alloc(256);
  float*       hscale = (float*)alloc((size_t)Mtok * 4);
  signed char* qh     = (signed char*)alloc((size_t)Mtok * Dff);
  float*       part   = (float*)alloc(4096 * 4);
  size_t phase = cur;
  _Float16* semn = (_Float16*)alloc((size_t)Mtok * Dsem * 2);
  _Float16* pron = (_Float16*)alloc((size_t)Mtok * Dpro * 2);
  _Float16* wq16 = (_Float16*)alloc((size_t)Dsem * Dsem * 2);
  _Float16* wk16 = (_Float16*)alloc((size_t)Dsem * Dpro * 2);
  _Float16* wv16 = (_Float16*)alloc((size_t)Dsem * Dpro * 2);
  _Float16* wo16 = (_Float16*)alloc((size_t)Dsem * Dsem * 2);
  _Float16* Qb   = (_Float16*)alloc((size_t)Mtok * Dsem * 2);
  _Float16* Kb   = (_Float16*)alloc((size_t)Mtok * Dsem * 2);
  _Float16* Vt   = (_Float16*)alloc((size_t)Mtok * Dsem * 2);
  _Float16* ctxb = (_Float16*)alloc((size_t)Mtok * Dsem * 2);
  float* hbuf = (float*)(wsb + phase);

  dim3 blk(256);
  rmsnorm_f16_kernel<<<Mtok, blk, 0, stream>>>(sem, g_sem, semn, Dsem);
  rmsnorm_f16_kernel<<<Mtok, blk, 0, stream>>>(pro, g_pro, pron, Dpro);
  f32_to_f16_kernel<<<(Dsem * Dsem) / 256, blk, 0, stream>>>(Wq, wq16, Dsem * Dsem);
  f32_to_f16_kernel<<<(Dsem * Dpro) / 256, blk, 0, stream>>>(Wk, wk16, Dsem * Dpro);
  f32_to_f16_kernel<<<(Dsem * Dpro) / 256, blk, 0, stream>>>(Wv, wv16, Dsem * Dpro);
  f32_to_f16_kernel<<<(Dsem * Dsem) / 256, blk, 0, stream>>>(Wo, wo16, Dsem * Dsem);
  int gqkv = (Mtok / 16) * (Dsem / 64) / 8;
  gemm_f16_kernel<<<gqkv, blk, 0, stream>>>(semn, wq16, bq, Qb, Dsem, Dsem, 0);
  gemm_f16_kernel<<<gqkv, blk, 0, stream>>>(pron, wk16, bk, Kb, Dsem, Dpro, 0);
  gemm_f16_kernel<<<gqkv, blk, 0, stream>>>(pron, wv16, bv, Vt, Dsem, Dpro, 1);
  attn_kernel<<<(Bb * Hh * (Ss / 16)) / 8, blk, 0, stream>>>(Qb, Kb, Vt, ctxb);
  gemm_o_kernel<<<gqkv, blk, 0, stream>>>(ctxb, wo16, bo, sem, sem2);
  xquant_kernel<<<Mtok, blk, 0, stream>>>(sem2, g_ff, qx, xscale);
  absmean_partial_kernel<<<1024, blk, 0, stream>>>(W1, Dff * Dsem, part);
  absmean_final_kernel<<<1, blk, 0, stream>>>(part, 1024, Dff * Dsem, w1s);
  absmean_partial_kernel<<<1024, blk, 0, stream>>>(W2, Dsem * Dff, part);
  absmean_final_kernel<<<1, blk, 0, stream>>>(part, 1024, Dsem * Dff, w2s);
  wquant_kernel<<<(Dff * Dsem) / 256, blk, 0, stream>>>(W1, w1s, qw1, Dff * Dsem);
  wquant_kernel<<<(Dsem * Dff) / 256, blk, 0, stream>>>(W2, w2s, qw2, Dsem * Dff);
  ffn1_kernel<<<(Mtok / 16) * (Dff / 64) / 8, blk, 0, stream>>>(qx, qw1, xscale, w1s,
                                                                alpha, beta, hbuf);
  hquant_kernel<<<Mtok, blk, 0, stream>>>(hbuf, qh, hscale);
  ffn2_kernel<<<(Mtok / 16) * (Dsem / 64) / 8, blk, 0, stream>>>(qh, qw2, hscale, w2s,
                                                                 sem2, outp);
}
